// AgentTwo_5394478923881
// MI455X (gfx1250) — matrix operand order repaired and checked by reading, not compiled, not run on hardware
//
#include <hip/hip_runtime.h>
#include <hip/hip_fp16.h>

typedef __attribute__((ext_vector_type(16))) _Float16 v16h;
typedef __attribute__((ext_vector_type(8)))  _Float16 v8h;
typedef __attribute__((ext_vector_type(8)))  float    v8f;
typedef __attribute__((ext_vector_type(4)))  float    f4;

namespace {
constexpr int kT       = 64;    // timesteps
constexpr int kE       = 128;   // hidden/embedding dim
constexpr int kA       = 10;    // actions
constexpr int k3E      = 384;   // 3*E gate rows
constexpr int kWStride = 136;   // padded LDS row stride (halves) for weights
constexpr int kHStride = 136;   // padded LDS row stride (halves) for h staging
constexpr int kWaves   = 4;
constexpr int kThreads = kWaves * 32;       // 128 threads = 4 waves
constexpr int kRowsPerBlock = kWaves * 16;  // 64 batch rows / block

constexpr size_t kWihHalves  = (size_t)k3E * kWStride;   // per weight matrix
constexpr size_t kHstgHalves = (size_t)16 * kHStride;    // per wave
constexpr size_t kLdsBytes =
    2 * kWihHalves * 2 +            // w_ih + w_hh as f16
    kWaves * kHstgHalves * 2 +      // per-wave h staging (f16)
    kWaves * 16 * kE * 4 +          // per-wave out staging (f32)
    kWaves * 16 * 4 +               // per-wave alive flags (f32)
    4 * kE * 4;                     // combined bias table (f32)
}

// B-matrix tile (32x16 f16, K x N) for wmma: weight matrix w stored row-major
// [out_col][k] in LDS; B[k][n] = w[colbase+n][kb+k].
// Lane layout: lanes 0-15 -> N=lane, K=0..7,16..23; lanes 16-31 -> N=lane-16, K=8..15,24..31.
__device__ __forceinline__ v16h load_b_tile(const _Float16* __restrict__ w,
                                            int colbase, int kb, int n, int hi) {
  const _Float16* p = w + (size_t)(colbase + n) * kWStride + kb + (hi ? 8 : 0);
  v8h b0 = *(const v8h*)p;
  v8h b1 = *(const v8h*)(p + 16);
  v16h b;
#pragma unroll
  for (int i = 0; i < 8; ++i) { b[i] = b0[i]; b[8 + i] = b1[i]; }
  return b;
}

__device__ __forceinline__ float fast_sigmoid(float x) {
  return 1.0f / (1.0f + __expf(-x));
}
__device__ __forceinline__ float fast_tanh(float x) {
  float e = __expf(-2.0f * x);
  return (1.0f - e) / (1.0f + e);
}

__global__ __launch_bounds__(kThreads, 1)
void gru_decode_kernel(const int*   __restrict__ utt,     // [B, T]
                       const float* __restrict__ emb,     // [V+1, E]
                       const float* __restrict__ w_ih,    // [3E, E]
                       const float* __restrict__ w_hh,    // [3E, E]
                       const float* __restrict__ b_ih,    // [3E]
                       const float* __restrict__ b_hh,    // [3E]
                       const float* __restrict__ w_out,   // [A, E]
                       const float* __restrict__ b_out,   // [A]
                       float*       __restrict__ out)     // [B, A]
{
  extern __shared__ char smem[];
  _Float16* wih_l     = (_Float16*)smem;
  _Float16* whh_l     = wih_l + kWihHalves;
  _Float16* hstg_all  = whh_l + kWihHalves;
  float*    ostg_all  = (float*)(hstg_all + kWaves * kHstgHalves);
  float*    alive_all = ostg_all + kWaves * 16 * kE;
  float*    bias_l    = alive_all + kWaves * 16;   // [4][kE]: r, z, in, hn

  const int tid  = threadIdx.x;
  const int lane = tid & 31;
  const int wave = tid >> 5;
  const int m    = lane & 15;           // row-within-tile for A / col for B,C
  const int hi   = (lane >> 4) & 1;     // lane half

  _Float16* hstg  = hstg_all + (size_t)wave * kHstgHalves;
  float*    ostg  = ostg_all + (size_t)wave * 16 * kE;
  float*    alive = alive_all + wave * 16;

  // ---- cooperative preload: weights -> LDS f16; biases; zero h; alive=1 ----
  for (int idx = tid; idx < k3E * kE; idx += kThreads) {
    int r = idx >> 7, c = idx & 127;
    wih_l[(size_t)r * kWStride + c] = (_Float16)w_ih[idx];
    whh_l[(size_t)r * kWStride + c] = (_Float16)w_hh[idx];
  }
  for (int c = tid; c < kE; c += kThreads) {
    bias_l[c]          = b_ih[c]          + b_hh[c];
    bias_l[kE + c]     = b_ih[kE + c]     + b_hh[kE + c];
    bias_l[2 * kE + c] = b_ih[2 * kE + c];
    bias_l[3 * kE + c] = b_hh[2 * kE + c];
  }
  for (int i = lane; i < 16 * kHStride; i += 32) hstg[i] = (_Float16)0.0f;
  if (lane < 16) alive[lane] = 1.0f;
  __syncthreads();

  const int myrow  = blockIdx.x * kRowsPerBlock + wave * 16 + m;
  const int koff_a = hi ? 8 : 0;   // A-layout K sub-offset per lane half

#pragma unroll 1
  for (int t = 0; t < kT; ++t) {
    const int tok = utt[myrow * kT + t];
    const float* erow = emb + (size_t)tok * kE;

    // alive flags entering this step (C-layout rows v / v+8) + own row
    float keep_v[8];
#pragma unroll
    for (int v = 0; v < 8; ++v) keep_v[v] = alive[hi ? (v + 8) : v];
    const float myAlive = alive[m];

    // ---- build A operands: emb (global gather, f32->f16) and h (LDS) ----
    v16h embA[4], hA[4];
#pragma unroll
    for (int kc = 0; kc < 4; ++kc) {
      const int kb = kc * 32;
      f4 s0 = *(const f4*)(erow + kb + koff_a);
      f4 s1 = *(const f4*)(erow + kb + koff_a + 4);
      f4 s2 = *(const f4*)(erow + kb + 16 + koff_a);
      f4 s3 = *(const f4*)(erow + kb + 16 + koff_a + 4);
      v16h a;
#pragma unroll
      for (int i = 0; i < 4; ++i) {
        a[i]      = (_Float16)s0[i];
        a[4 + i]  = (_Float16)s1[i];
        a[8 + i]  = (_Float16)s2[i];
        a[12 + i] = (_Float16)s3[i];
      }
      embA[kc] = a;

      const _Float16* hr = hstg + (size_t)m * kHStride;
      v8h h0 = *(const v8h*)(hr + kb + koff_a);
      v8h h1 = *(const v8h*)(hr + kb + 16 + koff_a);
      v16h hh;
#pragma unroll
      for (int i = 0; i < 8; ++i) { hh[i] = h0[i]; hh[8 + i] = h1[i]; }
      hA[kc] = hh;
    }

    // ---- 8 column chunks of E: r,z fused accumulate; n split ----
#pragma unroll
    for (int j = 0; j < 8; ++j) {
      const int col = j * 16 + m;

      // previous h in C layout (for z*h term), from staging
      float hold[8];
#pragma unroll
      for (int v = 0; v < 8; ++v) {
        int mv = hi ? (v + 8) : v;
        hold[v] = (float)hstg[(size_t)mv * kHStride + col];
      }

      const float br = bias_l[col];
      const float bz = bias_l[kE + col];
      const float bn = bias_l[2 * kE + col];
      const float bh = bias_l[3 * kE + col];

      v8f accR, accZ, accN, accH;
#pragma unroll
      for (int v = 0; v < 8; ++v) {
        accR[v] = br; accZ[v] = bz; accN[v] = bn; accH[v] = bh;
      }

#pragma unroll
      for (int kc = 0; kc < 4; ++kc) {
        const int kb = kc * 32;
        accR = __builtin_amdgcn_wmma_f32_16x16x32_f16(false, embA[kc], false,
                 load_b_tile(wih_l, j * 16,          kb, m, hi), (short)0, accR, false, false);
        accZ = __builtin_amdgcn_wmma_f32_16x16x32_f16(false, embA[kc], false,
                 load_b_tile(wih_l, kE + j * 16,     kb, m, hi), (short)0, accZ, false, false);
        accN = __builtin_amdgcn_wmma_f32_16x16x32_f16(false, embA[kc], false,
                 load_b_tile(wih_l, 2 * kE + j * 16, kb, m, hi), (short)0, accN, false, false);
      }
#pragma unroll
      for (int kc = 0; kc < 4; ++kc) {
        const int kb = kc * 32;
        accR = __builtin_amdgcn_wmma_f32_16x16x32_f16(false, hA[kc], false,
                 load_b_tile(whh_l, j * 16,          kb, m, hi), (short)0, accR, false, false);
        accZ = __builtin_amdgcn_wmma_f32_16x16x32_f16(false, hA[kc], false,
                 load_b_tile(whh_l, kE + j * 16,     kb, m, hi), (short)0, accZ, false, false);
        accH = __builtin_amdgcn_wmma_f32_16x16x32_f16(false, hA[kc], false,
                 load_b_tile(whh_l, 2 * kE + j * 16, kb, m, hi), (short)0, accH, false, false);
      }

      // ---- gates, h update, masked out_state write, restage h_new ----
#pragma unroll
      for (int v = 0; v < 8; ++v) {
        float r = fast_sigmoid(accR[v]);
        float z = fast_sigmoid(accZ[v]);
        float n = fast_tanh(accN[v] + r * accH[v]);
        float hnew = (1.0f - z) * n + z * hold[v];
        int mv = hi ? (v + 8) : v;
        if (keep_v[v] > 0.5f) ostg[mv * kE + col] = hnew;  // out_state while alive
        hstg[(size_t)mv * kHStride + col] = (_Float16)hnew;
      }
    }

    // mark_dead AFTER writing out_state
    if (lane < 16) alive[m] = (tok != 0) ? myAlive : 0.0f;
  }

  // ---- 128->10 head + softmax per row (out_state already in LDS) ----
  if (lane < 16) {
    const float* orow = ostg + m * kE;
    float lg[kA];
    float mx = -1e30f;
#pragma unroll
    for (int a = 0; a < kA; ++a) {
      const float* wr = w_out + a * kE;
      float s = b_out[a];
      for (int k = 0; k < kE; k += 4) {
        f4 o = *(const f4*)(orow + k);
        f4 w = *(const f4*)(wr + k);
        s += o[0] * w[0] + o[1] * w[1] + o[2] * w[2] + o[3] * w[3];
      }
      lg[a] = s;
      mx = fmaxf(mx, s);
    }
    float sum = 0.0f;
#pragma unroll
    for (int a = 0; a < kA; ++a) { lg[a] = __expf(lg[a] - mx); sum += lg[a]; }
    float inv = 1.0f / sum;
#pragma unroll
    for (int a = 0; a < kA; ++a) out[myrow * kA + a] = lg[a] * inv;
  }
}

extern "C" void kernel_launch(void* const* d_in, const int* in_sizes, int n_in,
                              void* d_out, int out_size, void* d_ws, size_t ws_size,
                              hipStream_t stream) {
  (void)in_sizes; (void)n_in; (void)out_size; (void)d_ws; (void)ws_size;
  const int*   utt   = (const int*)d_in[0];
  // d_in[1] = global_idxes (arange(B)) — unused by the reference math
  const float* emb   = (const float*)d_in[2];
  const float* w_ih  = (const float*)d_in[3];
  const float* w_hh  = (const float*)d_in[4];
  const float* b_ih  = (const float*)d_in[5];
  const float* b_hh  = (const float*)d_in[6];
  const float* w_out = (const float*)d_in[7];
  const float* b_out = (const float*)d_in[8];
  float* out = (float*)d_out;

  const int nBlocks = 8192 / kRowsPerBlock;  // 128

  // >64KB dynamic LDS: raise the limit (idempotent, deterministic).
  hipFuncSetAttribute(reinterpret_cast<const void*>(&gru_decode_kernel),
                      hipFuncAttributeMaxDynamicSharedMemorySize,
                      (int)kLdsBytes);

  gru_decode_kernel<<<nBlocks, kThreads, kLdsBytes, stream>>>(
      utt, emb, w_ih, w_hh, b_ih, b_hh, w_out, b_out, out);
}
